// TGCN_v2_78159814853224
// MI455X (gfx1250) — matrix operand order repaired and checked by reading, not compiled, not run on hardware
//
#include <hip/hip_runtime.h>

// ---- CDNA5 WMMA fragment types (wave32) ----
typedef __attribute__((ext_vector_type(16))) __bf16 v16bf;
typedef __attribute__((ext_vector_type(8)))  __bf16 v8bf;
typedef __attribute__((ext_vector_type(8)))  float  v8f;

#define BDIM 16
#define TDIM 128
#define VDIM 137
#define CDIM 3
#define VC   411        // V*C channels for data_bn
#define VP   160        // V padded to multiple of 32 (K dim of WMMA)
#define K9   9
#define PADK 4
#define NCLS 226
#define NT   (VP / 16)  // 10 N-tiles
#define KC   (VP / 32)  // 5 K-chunks of 32
#define BSTRIDE 40      // LDS B-chunk stride in elements (32 data + 8 pad = 80B)

union frag_u { v16bf v; v8bf h[2]; };

// ----------------------------------------------------------------------------
// BatchNorm1d statistics -> per-channel scale/shift
// ----------------------------------------------------------------------------
__global__ __launch_bounds__(256) void bn_stats_kernel(
    const float* __restrict__ X, const float* __restrict__ gamma,
    const float* __restrict__ beta, float* __restrict__ scsh) {
  __shared__ float s1[256];
  __shared__ float s2[256];
  const int ch = blockIdx.x;
  const int tid = threadIdx.x;
  float sum = 0.f, sq = 0.f;
  for (int i = tid; i < BDIM * TDIM; i += 256) {
    float x = X[(size_t)i * VC + ch];   // X = [B*T, 411] row-major
    sum += x; sq += x * x;
  }
  s1[tid] = sum; s2[tid] = sq;
  __syncthreads();
  for (int s = 128; s > 0; s >>= 1) {
    if (tid < s) { s1[tid] += s1[tid + s]; s2[tid] += s2[tid + s]; }
    __syncthreads();
  }
  if (tid == 0) {
    const float n = (float)(BDIM * TDIM);
    float mean = s1[0] / n;
    float var  = s2[0] / n - mean * mean;
    float sc   = gamma[ch] * rsqrtf(var + 1e-5f);
    scsh[2 * ch + 0] = sc;
    scsh[2 * ch + 1] = beta[ch] - mean * sc;
  }
}

// ----------------------------------------------------------------------------
// Pack conv weights (oc, ic, 9, 1) f32 -> Wt[k][oc][ic] bf16 (K=ic contiguous)
// ----------------------------------------------------------------------------
__global__ __launch_bounds__(256) void wt_pack_kernel(
    const float* __restrict__ w, __bf16* __restrict__ wt, int OC, int IC) {
  int e = blockIdx.x * 256 + threadIdx.x;
  if (e >= OC * IC * K9) return;
  int k  = e / (OC * IC);
  int r  = e % (OC * IC);
  int oc = r / IC, ic = r % IC;
  wt[e] = (__bf16)w[((size_t)oc * IC + ic) * K9 + k];
}

// ----------------------------------------------------------------------------
// conv0: BN-normalize on the fly + direct conv (3 -> 64). Planar bf16 out.
// ----------------------------------------------------------------------------
__global__ __launch_bounds__(256) void conv0_kernel(
    const float* __restrict__ X, const float* __restrict__ w0,
    const float* __restrict__ b0, const float* __restrict__ scsh,
    __bf16* __restrict__ out) {
  size_t e = (size_t)blockIdx.x * 256 + threadIdx.x;
  int v = (int)(e % VP); size_t r = e / VP;
  int t = (int)(r % TDIM); r /= TDIM;
  int oc = (int)(r % 64);
  int b  = (int)(r / 64);
  float val = 0.f;
  if (v < VDIM) {
    val = b0[oc];
    #pragma unroll
    for (int c = 0; c < CDIM; ++c) {
      const float sc = scsh[2 * (v * CDIM + c) + 0];
      const float sh = scsh[2 * (v * CDIM + c) + 1];
      #pragma unroll
      for (int k = 0; k < K9; ++k) {
        int ts = t + k - PADK;
        if (ts < 0 || ts >= TDIM) continue;
        float x = X[(((size_t)b * TDIM + ts) * VDIM + v) * CDIM + c] * sc + sh;
        val += w0[(oc * CDIM + c) * K9 + k] * x;
      }
    }
  }
  out[e] = (__bf16)val;
}

// ----------------------------------------------------------------------------
// Graph matmul: ax[ch,w] = sum_v x[ch,v] * A[v,w] per (b,t).
// Adjacency built straight into WMMA-B fragment layout in LDS:
//   Blds[(kk*VP + n)*40 + h*16 + j]  <=>  A[kk*32 + h*16 + j][n]
// -> B-fragment = 2x ds_load_b128, bank-conflict-free (80B lane stride).
// Output ax written in conv-fragment-ready layout F[b][ic/32][t][v][32]
// -> epilogue is ONE 16B store per lane.
// ----------------------------------------------------------------------------
__global__ __launch_bounds__(256) void graph_wmma_kernel(
    const float* __restrict__ X, const __bf16* __restrict__ xin,
    __bf16* __restrict__ F, int CH) {
  __shared__ __attribute__((aligned(16))) __bf16 Blds[KC * VP * BSTRIDE]; // 64000 B
  __shared__ __bf16 Xs[VC + 1];
  const int bt  = blockIdx.x;          // 0..2047
  const int b   = bt / TDIM, t = bt % TDIM;
  const int tid = threadIdx.x;

  // Load 137x3 joint slice (contiguous 411 floats -> bf16)
  for (int i = tid; i < VC; i += 256) Xs[i] = (__bf16)X[(size_t)bt * VC + i];
  __syncthreads();

  // Build padded adjacency directly in B-fragment layout:
  // exp(-dist/(2*delta^2)) = exp(-2*dist), delta = 0.5
  for (int e = tid; e < VP * VP; e += 256) {
    int v = e / VP, w = e % VP;
    float val = 0.f;
    if (v < VDIM && w < VDIM) {
      float dx = (float)Xs[v * 3 + 0] - (float)Xs[w * 3 + 0];
      float dy = (float)Xs[v * 3 + 1] - (float)Xs[w * 3 + 1];
      float dz = (float)Xs[v * 3 + 2] - (float)Xs[w * 3 + 2];
      val = expf(-2.0f * sqrtf(dx * dx + dy * dy + dz * dz));
    }
    Blds[(((v >> 5) * VP + w) * BSTRIDE) + ((v >> 4) & 1) * 16 + (v & 15)] =
        (__bf16)val;
  }
  __syncthreads();

  const int wi   = tid >> 5;
  const int lane = tid & 31;
  const int half = lane >> 4;
  const int col  = lane & 15;
  const int mtiles = CH >> 4;
  const int kch   = CH >> 5;           // channel 32-groups in F

  for (int tile = wi; tile < mtiles * NT; tile += 8) {
    const int m0 = (tile / NT) << 4;
    const int n0 = (tile % NT) << 4;
    v8f acc = {};
    #pragma unroll
    for (int kk = 0; kk < KC; ++kk) {
      // A fragment: x[b][m0+row][t][K..] planar, K contiguous along v
      const __bf16* arow =
          xin + (((size_t)b * CH + m0 + col) * TDIM + t) * VP + kk * 32;
      frag_u af;
      af.h[0] = *(const v8bf*)(arow + half * 16);
      af.h[1] = *(const v8bf*)(arow + half * 16 + 8);
      // B fragment: 2x 16B LDS loads, already in fragment order
      const __bf16* bp = &Blds[((kk * VP + n0 + col) * BSTRIDE) + half * 16];
      frag_u bf;
      bf.h[0] = *(const v8bf*)(bp);
      bf.h[1] = *(const v8bf*)(bp + 8);
      acc = __builtin_amdgcn_wmma_f32_16x16x32_bf16(
          false, af.v, false, bf.v, (short)0, acc, false, false);
    }
    // C fragment rows m0+half*8+i are 8 consecutive channels -> contiguous
    // inner offsets in F -> single 16B store per lane.
    v8bf y;
    #pragma unroll
    for (int i = 0; i < 8; ++i) y[i] = (__bf16)acc[i];
    const int ch0 = m0 + half * 8;
    __bf16* op = F +
        ((((size_t)b * kch + (ch0 >> 5)) * TDIM + t) * VP + n0 + col) * 32 +
        (ch0 & 31);
    *(v8bf*)op = y;
  }
}

// ----------------------------------------------------------------------------
// Temporal conv: y[oc,t,v] = relu(bias + sum_k W_k @ ax[:, t+k-4, v])
// B-fragments come from the fragment-ready F layout: 2x coalesced 16B loads.
// Templated on IC so the K-chunk loop fully unrolls.
// ----------------------------------------------------------------------------
template <int IC>
__global__ __launch_bounds__(256) void conv_wmma_kernel(
    const __bf16* __restrict__ F, const __bf16* __restrict__ wt,
    const float* __restrict__ bias, __bf16* __restrict__ out, int OC) {
  const int tid  = threadIdx.x;
  const int lane = tid & 31, half = lane >> 4, col = lane & 15;
  int gw = blockIdx.x * 8 + (tid >> 5);
  const int MT = OC >> 4;
  const int vt = gw % NT;  gw /= NT;
  const int t  = gw % TDIM; gw /= TDIM;
  const int mt = gw % MT;
  const int b  = gw / MT;
  const int m0 = mt << 4, n0 = vt << 4;

  v8f acc = {};
  for (int k = 0; k < K9; ++k) {
    const int ts = t + k - PADK;
    if (ts < 0 || ts >= TDIM) continue;   // uniform per wave: EXEC stays full
    #pragma unroll
    for (int kc = 0; kc < (IC >> 5); ++kc) {
      // A fragment: Wt[k][m0+row][kc*32..], K=ic contiguous (2x 16B loads)
      const __bf16* arow = wt + ((size_t)(k * OC + m0 + col)) * IC + kc * 32;
      frag_u af;
      af.h[0] = *(const v8bf*)(arow + half * 16);
      af.h[1] = *(const v8bf*)(arow + half * 16 + 8);
      // B fragment: F[b][kc][ts][n0+col][h*16 + j] -- contiguous 32B per lane
      const __bf16* bp = F +
          ((((size_t)b * (IC >> 5) + kc) * TDIM + ts) * VP + n0 + col) * 32 +
          half * 16;
      frag_u bf;
      bf.h[0] = *(const v8bf*)(bp);
      bf.h[1] = *(const v8bf*)(bp + 8);
      acc = __builtin_amdgcn_wmma_f32_16x16x32_bf16(
          false, af.v, false, bf.v, (short)0, acc, false, false);
    }
  }
  // Epilogue: bias + ReLU, zero v-padding; planar output for next graph stage
  #pragma unroll
  for (int i = 0; i < 8; ++i) {
    int ch = m0 + i + half * 8;
    int v  = n0 + col;
    float val = acc[i] + bias[ch];
    val = val > 0.f ? val : 0.f;
    if (v >= VDIM) val = 0.f;
    out[(((size_t)b * OC + ch) * TDIM + t) * VP + v] = (__bf16)val;
  }
}

// ----------------------------------------------------------------------------
// Global average pool over (T, V=137) -> pooled[b][256] f32
// ----------------------------------------------------------------------------
__global__ __launch_bounds__(256) void pool_kernel(
    const __bf16* __restrict__ x, float* __restrict__ pooled) {
  __shared__ float s[256];
  const int b  = blockIdx.x >> 8;
  const int ch = blockIdx.x & 255;
  const int tid = threadIdx.x;
  const __bf16* p = x + ((size_t)b * 256 + ch) * TDIM * VP;
  float sum = 0.f;
  for (int i = tid; i < TDIM * VDIM; i += 256) {
    int t = i / VDIM, v = i % VDIM;
    sum += (float)p[t * VP + v];
  }
  s[tid] = sum;
  __syncthreads();
  for (int st = 128; st > 0; st >>= 1) {
    if (tid < st) s[tid] += s[tid + st];
    __syncthreads();
  }
  if (tid == 0) pooled[blockIdx.x] = s[0] / (float)(TDIM * VDIM);
}

// ----------------------------------------------------------------------------
// Final FC 256 -> 226 (+bias), f32 output
// ----------------------------------------------------------------------------
__global__ __launch_bounds__(256) void fc_kernel(
    const float* __restrict__ pooled, const float* __restrict__ fw,
    const float* __restrict__ fb, float* __restrict__ out) {
  const int b = blockIdx.x, n = threadIdx.x;
  if (n >= NCLS) return;
  float s = fb[n];
  #pragma unroll 8
  for (int c = 0; c < 256; ++c) s += pooled[b * 256 + c] * fw[n * 256 + c];
  out[b * NCLS + n] = s;
}

// ----------------------------------------------------------------------------
extern "C" void kernel_launch(void* const* d_in, const int* in_sizes, int n_in,
                              void* d_out, int out_size, void* d_ws, size_t ws_size,
                              hipStream_t stream) {
  (void)in_sizes; (void)n_in; (void)out_size; (void)ws_size;
  const float* X     = (const float*)d_in[0];
  const float* gamma = (const float*)d_in[1];
  const float* beta  = (const float*)d_in[2];
  const float* w0    = (const float*)d_in[3];
  const float* b0    = (const float*)d_in[4];
  const float* w1    = (const float*)d_in[5];
  const float* b1    = (const float*)d_in[6];
  const float* w2    = (const float*)d_in[7];
  const float* b2    = (const float*)d_in[8];
  const float* w3    = (const float*)d_in[9];
  const float* b3    = (const float*)d_in[10];
  const float* fcw   = (const float*)d_in[11];
  const float* fcb   = (const float*)d_in[12];

  char* ws = (char*)d_ws;
  size_t off = 0;
  auto alloc = [&](size_t bytes) -> char* {
    char* p = ws + off;
    off = (off + bytes + 255) & ~(size_t)255;
    return p;
  };
  float*  scsh   = (float*)alloc(VC * 2 * sizeof(float));
  float*  pooled = (float*)alloc(BDIM * 256 * sizeof(float));
  __bf16* wt1    = (__bf16*)alloc((size_t)K9 * 64  * 64  * 2);
  __bf16* wt2    = (__bf16*)alloc((size_t)K9 * 128 * 64  * 2);
  __bf16* wt3    = (__bf16*)alloc((size_t)K9 * 256 * 128 * 2);
  __bf16* Xa     = (__bf16*)alloc((size_t)BDIM * 256 * TDIM * VP * 2); // 168 MB
  __bf16* Xb     = (__bf16*)alloc((size_t)BDIM * 256 * TDIM * VP * 2); // 168 MB
  __bf16* FR     = (__bf16*)alloc((size_t)BDIM * 128 * TDIM * VP * 2); //  84 MB

  // 1) BN stats -> scale/shift
  bn_stats_kernel<<<VC, 256, 0, stream>>>(X, gamma, beta, scsh);

  // 2) Pack conv weights into [k][oc][ic] bf16
  wt_pack_kernel<<<(64  * 64  * K9 + 255) / 256, 256, 0, stream>>>(w1, wt1, 64,  64);
  wt_pack_kernel<<<(128 * 64  * K9 + 255) / 256, 256, 0, stream>>>(w2, wt2, 128, 64);
  wt_pack_kernel<<<(256 * 128 * K9 + 255) / 256, 256, 0, stream>>>(w3, wt3, 256, 128);

  // 3) conv0 (BN fused), x0 -> Xa planar [16,64,128,160] bf16
  conv0_kernel<<<(int)(((size_t)BDIM * 64 * TDIM * VP) / 256), 256, 0, stream>>>(
      X, w0, b0, scsh, Xa);

  // 4) Block 1: graph (64ch) -> FR ; conv 64->64 -> Xb
  graph_wmma_kernel<<<BDIM * TDIM, 256, 0, stream>>>(X, Xa, FR, 64);
  conv_wmma_kernel<64><<<BDIM * 4 * TDIM * NT / 8, 256, 0, stream>>>(FR, wt1, b1, Xb, 64);

  // 5) Block 2: graph (64ch) -> FR ; conv 64->128 -> Xa
  graph_wmma_kernel<<<BDIM * TDIM, 256, 0, stream>>>(X, Xb, FR, 64);
  conv_wmma_kernel<64><<<BDIM * 8 * TDIM * NT / 8, 256, 0, stream>>>(FR, wt2, b2, Xa, 128);

  // 6) Block 3: graph (128ch) -> FR ; conv 128->256 -> Xb
  graph_wmma_kernel<<<BDIM * TDIM, 256, 0, stream>>>(X, Xa, FR, 128);
  conv_wmma_kernel<128><<<BDIM * 16 * TDIM * NT / 8, 256, 0, stream>>>(FR, wt3, b3, Xb, 256);

  // 7) Pool + FC -> d_out [16, 226] f32
  pool_kernel<<<BDIM * 256, 256, 0, stream>>>(Xb, pooled);
  fc_kernel<<<BDIM, 256, 0, stream>>>(pooled, fcw, fcb, (float*)d_out);
}
